// PPI_GNN_Optimized_20856361189535
// MI455X (gfx1250) — matrix operand order repaired and checked by reading, not compiled, not run on hardware
//
#include <hip/hip_runtime.h>

// ---------------------------------------------------------------- constants
#define NN      20000            // nodes
#define NE      320000           // edges
#define NP      320000           // pred edges
#define EA      (NE + NN)        // edges + self loops
#define NEG_SLOPE 0.2f
#define BN_EPS    1e-5f
#define PCH     64000            // predictor row chunk (multiple of 16, divides NP)

typedef __attribute__((ext_vector_type(16))) _Float16 v16h;
typedef __attribute__((ext_vector_type(8)))  float    v8f;

union Frag { v16h h; unsigned int u[8]; };

static inline int cdiv(long a, long b) { return (int)((a + b - 1) / b); }

// ---------------------------------------------------------------- WMMA GEMM
// C[M,N] = A[M,K] * B[K,N]  (B given pre-transposed as Bt[N,K], both f16)
// epilogue: + bias[n]  (+ r1v[m]*r1r[n])  -> optional relu -> f32 and/or f16 out
// One wave computes one 16x16 tile; K stepped by 32 (v_wmma_f32_16x16x32_f16).
// A frag: lane m=l%16; lanes<16 k=0..15, lanes>=16 k=16..31 (matches B).
// B frag (same contiguous k order as A): lane n=l%16; lanes<16 k=0..15, lanes>=16 k=16..31.
__global__ __launch_bounds__(256) void wmma_gemm_f16(
    const _Float16* __restrict__ A, const _Float16* __restrict__ Bt,
    const float* __restrict__ bias, const float* __restrict__ r1v,
    const float* __restrict__ r1r,
    float* __restrict__ Cf, _Float16* __restrict__ Ch,
    int M, int N, int K, int do_relu)
{
  const int lane = threadIdx.x & 31;
  const int hi   = lane >> 4;                      // half-wave select
  const int tnc  = N >> 4;
  const int tile = blockIdx.x * 8 + (threadIdx.x >> 5);
  if (tile >= (M >> 4) * tnc) return;
  const int tm = tile / tnc, tn = tile - tm * tnc;
  const int row = (tm << 4) + (lane & 15);
  const int col = (tn << 4) + (lane & 15);
  const unsigned int* __restrict__ Au = (const unsigned int*)(A  + (size_t)row * K);
  const unsigned int* __restrict__ Bu = (const unsigned int*)(Bt + (size_t)col * K);
  v8f acc = {};
  for (int k0 = 0; k0 < K; k0 += 32) {
    Frag a, b;
    const int ka = (k0 >> 1) + hi * 8;
    const int kb = (k0 >> 1) + hi * 8;
    __builtin_prefetch((const void*)(Au + ka + 16), 0, 1);   // global_prefetch_b8
#pragma unroll
    for (int v = 0; v < 8; ++v) a.u[v] = Au[ka + v];
#pragma unroll
    for (int v = 0; v < 8; ++v) b.u[v] = Bu[kb + v];
    acc = __builtin_amdgcn_wmma_f32_16x16x32_f16(false, a.h, false, b.h,
                                                 (short)0, acc, false, false);
  }
  const float bn = bias ? bias[col] : 0.0f;
  const float rn = r1r ? r1r[col] : 0.0f;
#pragma unroll
  for (int r = 0; r < 8; ++r) {                    // C/D: vgpr r -> M = r + 8*hi
    const int m = (tm << 4) + r + hi * 8;
    float v = acc[r] + bn;
    if (r1v) v = fmaf(r1v[m], rn, v);
    if (do_relu) v = fmaxf(v, 0.0f);
    const size_t idx = (size_t)m * N + col;
    if (Cf) Cf[idx] = v;
    if (Ch) Ch[idx] = (_Float16)v;
  }
}

// ---------------------------------------------------------------- small utils
__global__ void fill_f32(float* p, float v, int n) {
  int t = blockIdx.x * 256 + threadIdx.x; if (t < n) p[t] = v;
}
__global__ void cast_f16(const float* s, _Float16* d, int n) {
  int t = blockIdx.x * 256 + threadIdx.x; if (t < n) d[t] = (_Float16)s[t];
}
// W[K,N] (f32, row stride N) -> Wt[N,K] (f16)
__global__ void transpose_f16(const float* W, _Float16* Wt, int K, int N) {
  int t = blockIdx.x * 256 + threadIdx.x; if (t >= N * K) return;
  int n = t / K, k = t - n * K;
  Wt[t] = (_Float16)W[(size_t)k * N + n];
}

__device__ inline void edge_sd(const int* __restrict__ ei, int e, int& s, int& d) {
  if (e < NE) { s = ei[e]; d = ei[NE + e]; } else { s = d = e - NE; }
}
__device__ inline void atomicMaxF(float* addr, float v) {
  if (v >= 0.0f) atomicMax((int*)addr, __float_as_int(v));
  else           atomicMin((unsigned int*)addr, (unsigned int)__float_as_int(v));
}

// ---------------------------------------------------------------- GATv2 edge ops
__global__ void gat_logits(const float* __restrict__ xl, const float* __restrict__ xr,
                           const float* __restrict__ att, const int* __restrict__ ei,
                           float* __restrict__ logit, float* __restrict__ mmax,
                           int H, int C)
{
  int t = blockIdx.x * 256 + threadIdx.x;
  if (t >= EA * H) return;
  int e = t / H, h = t - e * H;
  int s, d; edge_sd(ei, e, s, d);
  const int F = H * C;
  const float* pl = xl + (size_t)s * F + h * C;
  const float* pr = xr + (size_t)d * F + h * C;
  const float* pa = att + h * C;
  float acc = 0.0f;
  for (int c = 0; c < C; ++c) {
    float v = pl[c] + pr[c];
    v = v > 0.0f ? v : NEG_SLOPE * v;            // leaky_relu
    acc = fmaf(v, pa[c], acc);
  }
  logit[t] = acc;
  atomicMaxF(&mmax[d * H + h], acc);             // segment_max over dst
}

__global__ void gat_exp(const float* __restrict__ logit, const float* __restrict__ mmax,
                        const int* __restrict__ ei, float* __restrict__ aexp,
                        float* __restrict__ denom, int H)
{
  int t = blockIdx.x * 256 + threadIdx.x;
  if (t >= EA * H) return;
  int e = t / H, h = t - e * H;
  int s, d; edge_sd(ei, e, s, d); (void)s;
  float a = expf(logit[t] - mmax[d * H + h]);
  aexp[t] = a;
  atomicAdd(&denom[d * H + h], a);               // segment_sum
}

__global__ void gat_scatter(const float* __restrict__ xl, const float* __restrict__ aexp,
                            const float* __restrict__ denom, const int* __restrict__ ei,
                            float* __restrict__ agg, int H, int C)
{
  long t = (long)blockIdx.x * 256 + threadIdx.x;
  const int F = H * C;
  if (t >= (long)EA * F) return;
  int e = (int)(t / F), f = (int)(t - (long)e * F), h = f / C;
  int s, d; edge_sd(ei, e, s, d);
  float alpha = aexp[(size_t)e * H + h] / denom[d * H + h];
  atomicAdd(&agg[(size_t)d * F + f], alpha * xl[(size_t)s * F + f]);
}

// ---------------------------------------------------------------- BatchNorm
__global__ __launch_bounds__(256) void bn_stats(const float* __restrict__ X,
                                                float* mu, float* var, int Nr, int F)
{
  __shared__ float s1[256], s2[256];
  const int f = blockIdx.x;
  float a = 0.0f, b = 0.0f;
  for (int n = threadIdx.x; n < Nr; n += 256) {
    float v = X[(size_t)n * F + f]; a += v; b += v * v;
  }
  s1[threadIdx.x] = a; s2[threadIdx.x] = b; __syncthreads();
  for (int st = 128; st > 0; st >>= 1) {
    if ((int)threadIdx.x < st) { s1[threadIdx.x] += s1[threadIdx.x + st];
                                 s2[threadIdx.x] += s2[threadIdx.x + st]; }
    __syncthreads();
  }
  if (threadIdx.x == 0) {
    float m = s1[0] / Nr;
    mu[f] = m;
    var[f] = fmaxf(s2[0] / Nr - m * m, 0.0f);
  }
}

__global__ void bn_relu(const float* __restrict__ X, const float* __restrict__ mu,
                        const float* __restrict__ var, const float* __restrict__ g,
                        const float* __restrict__ be, float* __restrict__ Yf,
                        _Float16* __restrict__ Yh, int total, int F)
{
  int t = blockIdx.x * 256 + threadIdx.x;
  if (t >= total) return;
  int f = t % F;
  float v = (X[t] - mu[f]) * rsqrtf(var[f] + BN_EPS) * g[f] + be[f];
  v = fmaxf(v, 0.0f);
  Yf[t] = v; Yh[t] = (_Float16)v;
}

// ---------------------------------------------------------------- predictors
__global__ void build_ef(const _Float16* __restrict__ z, const int* __restrict__ ps,
                         const int* __restrict__ pd, const float* __restrict__ sc,
                         _Float16* __restrict__ ef, float* __restrict__ pert,
                         int p0, int cnt)
{
  int t = blockIdx.x * 256 + threadIdx.x;
  if (t >= cnt * 64) return;
  int p = t >> 6, j = t & 63;
  int a = ps[p0 + p], b = pd[p0 + p];
  int node = (j < 32) ? a : b;
  ef[t] = z[node * 32 + (j & 31)];
  if (j == 0) pert[p] = sc[a] * sc[b];
}

__global__ void mlp_final(const _Float16* __restrict__ h2, const float* __restrict__ w,
                          const float* __restrict__ b, float* __restrict__ out,
                          int p0, int cnt)
{
  int p = blockIdx.x * 256 + threadIdx.x;
  if (p >= cnt) return;
  const _Float16* hp = h2 + (size_t)p * 64;
  float a0 = b[0], a1 = b[1];
  for (int k = 0; k < 64; ++k) {
    float v = (float)hp[k];
    a0 = fmaf(v, w[k * 2 + 0], a0);
    a1 = fmaf(v, w[k * 2 + 1], a1);
  }
  out[(size_t)(p0 + p) * 2 + 0] = a0;
  out[(size_t)(p0 + p) * 2 + 1] = a1;
}

// ---------------------------------------------------------------- host driver
extern "C" void kernel_launch(void* const* d_in, const int* in_sizes, int n_in,
                              void* d_out, int out_size, void* d_ws, size_t ws_size,
                              hipStream_t stream)
{
  (void)in_sizes; (void)n_in; (void)out_size; (void)ws_size;
  const float* x      = (const float*)d_in[0];
  const int*   ei     = (const int*)d_in[1];
  const int*   pe     = (const int*)d_in[2];
  const float* scores = (const float*)d_in[3];
  const float* w1l = (const float*)d_in[4];  const float* w1r = (const float*)d_in[5];
  const float* att1= (const float*)d_in[6];
  const float* g1  = (const float*)d_in[8];  const float* be1 = (const float*)d_in[9];
  const float* w2l = (const float*)d_in[10]; const float* w2r = (const float*)d_in[11];
  const float* att2= (const float*)d_in[12];
  const float* g2  = (const float*)d_in[14]; const float* be2 = (const float*)d_in[15];
  const float* w3l = (const float*)d_in[16]; const float* w3r = (const float*)d_in[17];
  const float* att3= (const float*)d_in[18];
  const float* g3  = (const float*)d_in[20]; const float* be3 = (const float*)d_in[21];
  const float* epw1= (const float*)d_in[22]; const float* epb1= (const float*)d_in[23];
  const float* epw2= (const float*)d_in[24]; const float* epb2= (const float*)d_in[25];
  const float* epw3= (const float*)d_in[26]; const float* epb3= (const float*)d_in[27];
  const float* ppw1= (const float*)d_in[28]; const float* ppb1= (const float*)d_in[29];
  const float* ppw2= (const float*)d_in[30]; const float* ppb2= (const float*)d_in[31];
  const float* ppw3= (const float*)d_in[32]; const float* ppb3= (const float*)d_in[33];
  // (bias1/2/3 are cancelled by BatchNorm: per-channel shift -> no effect)

  // ---- workspace layout (~116 MB)
  char* ws = (char*)d_ws; size_t off = 0;
  auto take = [&](size_t bytes) -> char* {
    char* p = ws + off; off += (bytes + 255) & ~(size_t)255; return p;
  };
  _Float16* t_w1l = (_Float16*)take((size_t)256 * 128 * 2);
  _Float16* t_w1r = (_Float16*)take((size_t)256 * 128 * 2);
  _Float16* t_w2l = (_Float16*)take((size_t)256 * 256 * 2);
  _Float16* t_w2r = (_Float16*)take((size_t)256 * 256 * 2);
  _Float16* t_w3l = (_Float16*)take((size_t)32  * 256 * 2);
  _Float16* t_w3r = (_Float16*)take((size_t)32  * 256 * 2);
  _Float16* t_ep1 = (_Float16*)take((size_t)128 * 64  * 2);
  _Float16* t_ep2 = (_Float16*)take((size_t)64  * 128 * 2);
  _Float16* t_pp1 = (_Float16*)take((size_t)128 * 64  * 2);
  _Float16* t_pp2 = (_Float16*)take((size_t)64  * 128 * 2);
  float* mu    = (float*)take(256 * 4);
  float* var   = (float*)take(256 * 4);
  float* mmax  = (float*)take((size_t)NN * 8 * 4);
  float* denom = (float*)take((size_t)NN * 8 * 4);
  float*    cur   = (float*)   take((size_t)NN * 256 * 4);
  _Float16* cur16 = (_Float16*)take((size_t)NN * 256 * 2);
  float* xl    = (float*)take((size_t)NN * 256 * 4);
  float* xr    = (float*)take((size_t)NN * 256 * 4);
  float* agg   = (float*)take((size_t)NN * 256 * 4);
  float* logit = (float*)take((size_t)EA * 8 * 4);
  float* aexp  = (float*)take((size_t)EA * 8 * 4);
  // MLP chunk buffers alias GNN scratch (dead after z is produced)
  _Float16* ef16  = (_Float16*)xl;    // 64k x 64  f16 (8.2MB  <= 20.5MB)
  _Float16* mbuf1 = (_Float16*)xr;    // 64k x 128 f16 (16.4MB <= 20.5MB)
  _Float16* mbuf2 = (_Float16*)agg;   // 64k x 64  f16
  float*    pert  = logit;            // 64k f32

  auto T = [&](const float* W, _Float16* Wt, int K, int N) {
    transpose_f16<<<cdiv((long)N * K, 256), 256, 0, stream>>>(W, Wt, K, N);
  };
  T(w1l, t_w1l, 128, 256); T(w1r, t_w1r, 128, 256);
  T(w2l, t_w2l, 256, 256); T(w2r, t_w2r, 256, 256);
  T(w3l, t_w3l, 256, 32);  T(w3r, t_w3r, 256, 32);
  T(epw1, t_ep1, 64, 128); T(epw2, t_ep2, 128, 64);
  T(ppw1, t_pp1, 64, 128); /* rows 0..63 of [65,128]; row 64 via rank-1 */
  T(ppw2, t_pp2, 128, 64);

  // x -> f16 (layer-1 GEMM input)
  cast_f16<<<cdiv((long)NN * 128, 256), 256, 0, stream>>>(x, cur16, NN * 128);

  auto run_layer = [&](const _Float16* wlT, const _Float16* wrT, const float* att,
                       const float* g, const float* be, int Fin, int H, int C) {
    const int Fout = H * C;
    const int tiles = (NN / 16) * (Fout / 16);
    wmma_gemm_f16<<<cdiv(tiles, 8), 256, 0, stream>>>(
        cur16, wlT, nullptr, nullptr, nullptr, xl, nullptr, NN, Fout, Fin, 0);
    wmma_gemm_f16<<<cdiv(tiles, 8), 256, 0, stream>>>(
        cur16, wrT, nullptr, nullptr, nullptr, xr, nullptr, NN, Fout, Fin, 0);
    fill_f32<<<cdiv((long)NN * H, 256), 256, 0, stream>>>(mmax, -3.0e38f, NN * H);
    fill_f32<<<cdiv((long)NN * H, 256), 256, 0, stream>>>(denom, 0.0f, NN * H);
    fill_f32<<<cdiv((long)NN * Fout, 256), 256, 0, stream>>>(agg, 0.0f, NN * Fout);
    gat_logits<<<cdiv((long)EA * H, 256), 256, 0, stream>>>(xl, xr, att, ei, logit, mmax, H, C);
    gat_exp<<<cdiv((long)EA * H, 256), 256, 0, stream>>>(logit, mmax, ei, aexp, denom, H);
    gat_scatter<<<cdiv((long)EA * Fout, 256), 256, 0, stream>>>(xl, aexp, denom, ei, agg, H, C);
    bn_stats<<<Fout, 256, 0, stream>>>(agg, mu, var, NN, Fout);
    bn_relu<<<cdiv((long)NN * Fout, 256), 256, 0, stream>>>(
        agg, mu, var, g, be, cur, cur16, NN * Fout, Fout);
  };

  run_layer(t_w1l, t_w1r, att1, g1, be1, 128, 8, 32);   // 128 -> 8x32
  run_layer(t_w2l, t_w2r, att2, g2, be2, 256, 4, 64);   // 256 -> 4x64
  run_layer(t_w3l, t_w3r, att3, g3, be3, 256, 1, 32);   // 256 -> 1x32  => z in cur16

  // ---- edge predictors, chunked over P
  const int* ps = pe; const int* pd = pe + NP;
  float* out_n = (float*)d_out;
  float* out_p = out_n + (size_t)NP * 2;
  for (int p0 = 0; p0 < NP; p0 += PCH) {
    const int cnt = PCH;
    build_ef<<<cdiv((long)cnt * 64, 256), 256, 0, stream>>>(
        cur16, ps, pd, scores, ef16, pert, p0, cnt);
    const int t1 = (cnt / 16) * (128 / 16);
    const int t2 = (cnt / 16) * (64 / 16);
    // normal predictor: 64 -> 128 -> 64 -> 2
    wmma_gemm_f16<<<cdiv(t1, 8), 256, 0, stream>>>(
        ef16, t_ep1, epb1, nullptr, nullptr, nullptr, mbuf1, cnt, 128, 64, 1);
    wmma_gemm_f16<<<cdiv(t2, 8), 256, 0, stream>>>(
        mbuf1, t_ep2, epb2, nullptr, nullptr, nullptr, mbuf2, cnt, 64, 128, 1);
    mlp_final<<<cdiv(cnt, 256), 256, 0, stream>>>(mbuf2, epw3, epb3, out_n, p0, cnt);
    // perturbed predictor: 65 -> 128 -> 64 -> 2 ; column 65 as rank-1 update
    wmma_gemm_f16<<<cdiv(t1, 8), 256, 0, stream>>>(
        ef16, t_pp1, ppb1, pert, ppw1 + (size_t)64 * 128, nullptr, mbuf1, cnt, 128, 64, 1);
    wmma_gemm_f16<<<cdiv(t2, 8), 256, 0, stream>>>(
        mbuf1, t_pp2, ppb2, nullptr, nullptr, nullptr, mbuf2, cnt, 64, 128, 1);
    mlp_final<<<cdiv(cnt, 256), 256, 0, stream>>>(mbuf2, ppw3, ppb3, out_p, p0, cnt);
  }
}
